// WeakREChead_57174604644449
// MI455X (gfx1250) — matrix operand order repaired and checked by reading, not compiled, not run on hardware
//
#include <hip/hip_runtime.h>
#include <hip/hip_bf16.h>

typedef __attribute__((ext_vector_type(16))) __bf16         v16bf;
typedef __attribute__((ext_vector_type(16))) unsigned short v16us;
typedef __attribute__((ext_vector_type(8)))  unsigned short v8us;
typedef __attribute__((ext_vector_type(8)))  float          v8f;

constexpr int kB  = 64;    // batch
constexpr int kV  = 2048;  // visual anchors
constexpr int kVT = 2048;  // tag slots
constexpr int kD  = 256;   // embedding dim
constexpr int kVC = 128;   // v-chunk per workgroup
constexpr int kNC = kV / kVC;  // 16 chunks
constexpr int kRow = kVC + 1;  // padded LDS row stride (bank-conflict free)

// f32 -> bf16 via native cast (backend may select v_cvt_pk_bf16_f32)
__device__ __forceinline__ v16bf cvtB(const float* __restrict__ p) {
  v16bf r;
#pragma unroll
  for (int i = 0; i < 16; ++i) r[i] = (__bf16)p[i];
  return r;
}

// A fragment from pre-converted bf16: elems 0..7 at p, 8..15 at p+8
__device__ __forceinline__ v16bf loadA(const unsigned short* __restrict__ p) {
  v8us lo = *reinterpret_cast<const v8us*>(p);
  v8us hi = *reinterpret_cast<const v8us*>(p + 8);
  v16us u = __builtin_shufflevector(lo, hi, 0, 1, 2, 3, 4, 5, 6, 7,
                                    8, 9, 10, 11, 12, 13, 14, 15);
  return __builtin_bit_cast(v16bf, u);
}

// ---------------------------------------------------------------------------
// K0: convert lan [64,256] f32 -> bf16 (row-major ushort) once.
// ---------------------------------------------------------------------------
__global__ __launch_bounds__(256) void k_lan_cvt(const float* __restrict__ lan,
                                                 unsigned short* __restrict__ lanbf) {
  const int t = threadIdx.x;
#pragma unroll
  for (int i = 0; i < (kB * kD) / 256; ++i) {
    const int idx = i * 256 + t;
    lanbf[idx] = __builtin_bit_cast(unsigned short, (__bf16)lan[idx]);
  }
}

// ---------------------------------------------------------------------------
// K1: per (a, v-chunk): S[b, v] = lan[b] . vis[a, v] via bf16 WMMA,
//     then per-b top-2 over the 128 v's of this chunk.
// partial layout: [a][b][chunk][2]
// ---------------------------------------------------------------------------
__global__ __launch_bounds__(256) void k_vis_top2(
    const float* __restrict__ vis, const unsigned short* __restrict__ lanbf,
    float* __restrict__ partial) {
  __shared__ float S[kB * kRow];        // ~33 KB padded score tile
  __shared__ float P[kB * 4 * 2];       // per-quarter top-2 partials

  const int a     = blockIdx.x;
  const int chunk = blockIdx.y;
  const int wave  = threadIdx.x >> 5;
  const int lane  = threadIdx.x & 31;
  const int half  = lane >> 4;   // 0: lanes 0-15, 1: lanes 16-31
  const int l16   = lane & 15;

  // This wave's 16-v strip; B column n = vis row (vbase + n), contiguous in d.
  const int vbase = chunk * kVC + wave * 16;
  const float* colb =
      vis + ((size_t)a * kV + vbase + l16) * kD + (half ? 16 : 0);
  const unsigned short* lrow = lanbf + (size_t)l16 * kD + (half ? 16 : 0);

  v8f acc0 = {}, acc1 = {}, acc2 = {}, acc3 = {};

  for (int k = 0; k < kD; k += 32) {
    __builtin_prefetch(colb + k + 32, 0, 1);  // next K block of this column
    v16bf bf = cvtB(colb + k);                // 16 contiguous K per lane half

    v16bf a0 = loadA(lrow + k);
    v16bf a1 = loadA(lrow + 16 * kD + k);
    v16bf a2 = loadA(lrow + 32 * kD + k);
    v16bf a3 = loadA(lrow + 48 * kD + k);

    acc0 = __builtin_amdgcn_wmma_f32_16x16x32_bf16(false, a0, false, bf,
                                                   (short)0, acc0, false, false);
    acc1 = __builtin_amdgcn_wmma_f32_16x16x32_bf16(false, a1, false, bf,
                                                   (short)0, acc1, false, false);
    acc2 = __builtin_amdgcn_wmma_f32_16x16x32_bf16(false, a2, false, bf,
                                                   (short)0, acc2, false, false);
    acc3 = __builtin_amdgcn_wmma_f32_16x16x32_bf16(false, a3, false, bf,
                                                   (short)0, acc3, false, false);
  }

  // C layout: VGPR r, lanes 0-15: M=r ; lanes 16-31: M=8+r ; N = l16
  const int vloc = wave * 16 + l16;
#pragma unroll
  for (int r = 0; r < 8; ++r) {
    const int m = half * 8 + r;
    S[(0 * 16 + m) * kRow + vloc] = acc0[r];
    S[(1 * 16 + m) * kRow + vloc] = acc1[r];
    S[(2 * 16 + m) * kRow + vloc] = acc2[r];
    S[(3 * 16 + m) * kRow + vloc] = acc3[r];
  }
  __syncthreads();

  // stage 1: 4 threads per b-row, each scans 32 values
  {
    const int b = threadIdx.x >> 2, q = threadIdx.x & 3;
    const float* src = S + b * kRow + q * 32;
    float m1 = -3.4e38f, m2 = -3.4e38f;
#pragma unroll 4
    for (int v = 0; v < 32; ++v) {
      float x = src[v];
      if (x > m1) { m2 = m1; m1 = x; }
      else if (x > m2) { m2 = x; }
    }
    P[(b * 4 + q) * 2 + 0] = m1;
    P[(b * 4 + q) * 2 + 1] = m2;
  }
  __syncthreads();

  // stage 2: merge the 4 quarter top-2's per row
  if (threadIdx.x < kB) {
    const int b = threadIdx.x;
    float m1 = -3.4e38f, m2 = -3.4e38f;
#pragma unroll
    for (int c = 0; c < 8; ++c) {
      float x = P[b * 8 + c];
      if (x > m1) { m2 = m1; m1 = x; }
      else if (x > m2) { m2 = x; }
    }
    float* o = partial + (((size_t)a * kB + b) * kNC + chunk) * 2;
    o[0] = m1;
    o[1] = m2;
  }
}

// ---------------------------------------------------------------------------
// K2: merge chunk partials -> vl0[b][a] (max), vl1[b][a] (2nd max)
// ---------------------------------------------------------------------------
__global__ __launch_bounds__(256) void k_merge(const float* __restrict__ partial,
                                               float* __restrict__ vl0,
                                               float* __restrict__ vl1) {
  int p = blockIdx.x * 256 + threadIdx.x;
  if (p >= kB * kB) return;
  int a = p >> 6, b = p & 63;
  const float* src = partial + ((size_t)a * kB + b) * kNC * 2;
  float m1 = -3.4e38f, m2 = -3.4e38f;
#pragma unroll
  for (int c = 0; c < kNC * 2; ++c) {
    float x = src[c];
    if (x > m1) { m2 = m1; m1 = x; }
    else if (x > m2) { m2 = x; }
  }
  vl0[b * kB + a] = m1;
  vl1[b * kB + a] = m2;
}

// ---------------------------------------------------------------------------
// K3: only tag slots vt=0,1 ever reach the loss. Exact f32 dots.
// ---------------------------------------------------------------------------
__global__ __launch_bounds__(256) void k_tag(const float* __restrict__ lan,
                                             const float* __restrict__ tag,
                                             float* __restrict__ T0,
                                             float* __restrict__ T1) {
  const int t = threadIdx.x;       // 256 threads: b = t/4, 16 a's each
  const int b = t >> 2, sub = t & 3;
  const float* l = lan + (size_t)b * kD;
  for (int ai = 0; ai < 16; ++ai) {
    const int a = sub * 16 + ai;
    const float* g0 = tag + (size_t)a * kVT * kD;  // tag_fs[a,0,:]
    const float* g1 = g0 + kD;                     // tag_fs[a,1,:]
    float s0 = 0.f, s1 = 0.f;
#pragma unroll 4
    for (int d = 0; d < kD; ++d) {
      s0 = fmaf(l[d], g0[d], s0);
      s1 = fmaf(l[d], g1[d], s1);
    }
    T0[b * kB + a] = s0;
    T1[b * kB + a] = s1;
  }
}

// stable-descending rank (matches lax.top_k tie ordering): index with rank k
__device__ __forceinline__ int kth_idx(const float* x, int k) {
  for (int i = 0; i < kB; ++i) {
    float xi = x[i];
    int r = 0;
    for (int j = 0; j < kB; ++j) {
      float xj = x[j];
      if (xj > xi || (xj == xi && j < i)) ++r;
    }
    if (r == k) return i;
  }
  return 0;
}

// ---------------------------------------------------------------------------
// K4: per-row log-sum-exp losses; loss = mean_b(lv_b + lt_b)
// ---------------------------------------------------------------------------
__global__ __launch_bounds__(64) void k_loss(const float* __restrict__ vl0,
                                             const float* __restrict__ vl1,
                                             const float* __restrict__ T0,
                                             const float* __restrict__ T1,
                                             float* __restrict__ out) {
  __shared__ float red[kB];
  const int b = threadIdx.x;

  // ---- visual: logits row = vl0[b,:] ++ vl1[b, a != b]; target = vl0[b,b]
  const float* r0 = vl0 + b * kB;
  const float* r1 = vl1 + b * kB;
  float M = -3.4e38f;
  for (int a = 0; a < kB; ++a) {
    M = fmaxf(M, r0[a]);
    if (a != b) M = fmaxf(M, r1[a]);
  }
  float s = 0.f;
  for (int a = 0; a < kB; ++a) {
    s += expf(r0[a] - M);
    if (a != b) s += expf(r1[a] - M);
  }
  float lv = (M + logf(s)) - r0[b];

  // ---- tag: logits row = sort(T0 row) ++ (sort(T1 row) minus rank-b elem);
  //      target = rank-b of T0 row. LSE is order-invariant.
  const float* t0 = T0 + b * kB;
  const float* t1 = T1 + b * kB;
  const int i0 = kth_idx(t0, b);
  const int i1 = kth_idx(t1, b);
  const float tgt = t0[i0];
  float M2 = -3.4e38f;
  for (int a = 0; a < kB; ++a) {
    M2 = fmaxf(M2, t0[a]);
    if (a != i1) M2 = fmaxf(M2, t1[a]);
  }
  float s2 = 0.f;
  for (int a = 0; a < kB; ++a) {
    s2 += expf(t0[a] - M2);
    if (a != i1) s2 += expf(t1[a] - M2);
  }
  float lt = (M2 + logf(s2)) - tgt;

  red[b] = lv + lt;
  __syncthreads();
  if (b == 0) {
    float tot = 0.f;
    for (int i = 0; i < kB; ++i) tot += red[i];
    out[0] = tot / (float)kB;
  }
}

extern "C" void kernel_launch(void* const* d_in, const int* in_sizes, int n_in,
                              void* d_out, int out_size, void* d_ws, size_t ws_size,
                              hipStream_t stream) {
  const float* vis = (const float*)d_in[0];  // [64,2048,256]
  const float* lan = (const float*)d_in[1];  // [64,1,256]
  const float* tag = (const float*)d_in[2];  // [64,2048,256]
  float* out = (float*)d_out;                // scalar

  // workspace carving: partial 512KB, 4x 16KB f32 matrices, 32KB bf16 lan
  float* partial = (float*)d_ws;                       // [64][64][16][2]
  float* vl0 = partial + (size_t)kB * kB * kNC * 2;    // [b][a]
  float* vl1 = vl0 + kB * kB;
  float* T0  = vl1 + kB * kB;
  float* T1  = T0 + kB * kB;
  unsigned short* lanbf = (unsigned short*)(T1 + kB * kB);  // [64][256] bf16

  k_lan_cvt<<<1, 256, 0, stream>>>(lan, lanbf);
  dim3 g1(kB, kNC);
  k_vis_top2<<<g1, 256, 0, stream>>>(vis, lanbf, partial);
  k_merge<<<(kB * kB + 255) / 256, 256, 0, stream>>>(partial, vl0, vl1);
  k_tag<<<1, 256, 0, stream>>>(lan, tag, T0, T1);
  k_loss<<<1, kB, 0, stream>>>(vl0, vl1, T0, T1, out);
}
